// Conv_60172491817529
// MI455X (gfx1250) — matrix operand order repaired and checked by reading, not compiled, not run on hardware
//
#include <hip/hip_runtime.h>

// Sizes fixed by the reference
#define BATCH 512
#define MBS   32768
#define FLEN  512
#define CDIM  16
#define F2    1024

typedef __attribute__((ext_vector_type(16))) __bf16          v16bf;
typedef __attribute__((ext_vector_type(16))) unsigned short  v16u16;
typedef __attribute__((ext_vector_type(8)))  float           v8f;

#if __has_builtin(__builtin_amdgcn_global_load_async_to_lds_b128) && \
    __has_builtin(__builtin_amdgcn_s_wait_asynccnt)
#define HAVE_ASYNC_LDS 1
typedef int v4i_t __attribute__((vector_size(16)));
typedef __attribute__((address_space(1))) v4i_t* gv4i_p;   // global int4*
typedef __attribute__((address_space(3))) v4i_t* lv4i_p;   // LDS int4*
#endif

static __device__ __forceinline__ unsigned short f2bf(float f) {
    unsigned int u = __builtin_bit_cast(unsigned int, f);
    u += 0x7FFFu + ((u >> 16) & 1u);           // round-to-nearest-even
    return (unsigned short)(u >> 16);
}

struct B32B { uint4 lo, hi; };                 // 32-byte fragment carrier

// -------------------------------------------------------------------------
// Kernel 1: conditioning head  z = FiLM(PReLU(c@W0+b0))  -> bf16 workspace
// -------------------------------------------------------------------------
__global__ void cond_film_kernel(const float* __restrict__ c,
                                 const float* __restrict__ W0, const float* __restrict__ b0,
                                 const float* __restrict__ alpha,
                                 const float* __restrict__ Wg, const float* __restrict__ bg,
                                 const float* __restrict__ Wb, const float* __restrict__ bb,
                                 unsigned short* __restrict__ zbf) {
    int b = blockIdx.x;
    int f = threadIdx.x;                 // 512 threads
    __shared__ float cs[CDIM];
    if (threadIdx.x < CDIM) cs[threadIdx.x] = c[b * CDIM + threadIdx.x];
    __syncthreads();
    float d0 = 0.f, dg = 0.f, db = 0.f;
#pragma unroll
    for (int k = 0; k < CDIM; ++k) {
        float ck = cs[k];
        d0 += ck * W0[k * FLEN + f];
        dg += ck * Wg[k * FLEN + f];
        db += ck * Wb[k * FLEN + f];
    }
    float z0 = d0 + b0[f];
    z0 = (z0 >= 0.f) ? z0 : alpha[f] * z0;          // PReLU
    float z = z0 * (dg + bg[f]) + (db + bb[f]);     // FiLM
    zbf[b * FLEN + f] = f2bf(z);
}

// -------------------------------------------------------------------------
// Kernel 2: G[512x1024] = Zbf[512x512] @ Wglu[512x1024]   (WMMA bf16)
// -------------------------------------------------------------------------
__global__ void glu_gemm_kernel(const unsigned short* __restrict__ zbf,
                                const float* __restrict__ Wglu,
                                float* __restrict__ G) {
    int wave = (blockIdx.x * blockDim.x + threadIdx.x) >> 5;   // 2048 tiles
    int lane = threadIdx.x & 31;
    int tn = wave & 63, tm = wave >> 6;
    int b0r = tm * 16, n0 = tn * 16;
    int i = lane & 15, half = lane >> 4;
    v8f acc = {};
    for (int kc = 0; kc < 16; ++kc) {
        int kbase = kc * 32;
        v16u16 au, bu;
#pragma unroll
        for (int p = 0; p < 8; ++p) {
            int ka = kbase + 16 * (p >> 2) + half * 8 + 2 * (p & 3);
            unsigned int apair = *(const unsigned int*)(zbf + (b0r + i) * FLEN + ka);
            au[2 * p]     = (unsigned short)(apair & 0xFFFFu);
            au[2 * p + 1] = (unsigned short)(apair >> 16);
            int kb = kbase + 16 * (p >> 2) + half * 8 + 2 * (p & 3);
            bu[2 * p]     = f2bf(Wglu[(kb)     * F2 + n0 + i]);
            bu[2 * p + 1] = f2bf(Wglu[(kb + 1) * F2 + n0 + i]);
        }
        acc = __builtin_amdgcn_wmma_f32_16x16x32_bf16(
                  false, __builtin_bit_cast(v16bf, au),
                  false, __builtin_bit_cast(v16bf, bu),
                  (short)0, acc, false, false);
    }
#pragma unroll
    for (int v = 0; v < 8; ++v)
        G[(b0r + v + 8 * half) * F2 + n0 + i] = acc[v];
}

// -------------------------------------------------------------------------
// Kernel 3: GLU gate  zg = (a+bglu_a) * sigmoid(gate+bglu_g)  -> bf16
// -------------------------------------------------------------------------
__global__ void glu_act_kernel(const float* __restrict__ G,
                               const float* __restrict__ bglu,
                               unsigned short* __restrict__ zgbf) {
    int idx = blockIdx.x * blockDim.x + threadIdx.x;   // 512*512
    int b = idx >> 9, f = idx & 511;
    float a = G[b * F2 + f] + bglu[f];
    float g = G[b * F2 + FLEN + f] + bglu[FLEN + f];
    float s = 1.f / (1.f + __expf(-g));
    zgbf[idx] = f2bf(a * s);
}

// -------------------------------------------------------------------------
// Kernel 4: H = softsign(ZG @ Wf + bf)  [512x512]  (WMMA bf16) -> bf16
// -------------------------------------------------------------------------
__global__ void filt_gemm_kernel(const unsigned short* __restrict__ zgbf,
                                 const float* __restrict__ Wf,
                                 const float* __restrict__ bfv,
                                 unsigned short* __restrict__ hbf) {
    int wave = (blockIdx.x * blockDim.x + threadIdx.x) >> 5;   // 1024 tiles
    int lane = threadIdx.x & 31;
    int tn = wave & 31, tm = wave >> 5;
    int b0r = tm * 16, n0 = tn * 16;
    int i = lane & 15, half = lane >> 4;
    v8f acc = {};
    for (int kc = 0; kc < 16; ++kc) {
        int kbase = kc * 32;
        v16u16 au, bu;
#pragma unroll
        for (int p = 0; p < 8; ++p) {
            int ka = kbase + 16 * (p >> 2) + half * 8 + 2 * (p & 3);
            unsigned int apair = *(const unsigned int*)(zgbf + (b0r + i) * FLEN + ka);
            au[2 * p]     = (unsigned short)(apair & 0xFFFFu);
            au[2 * p + 1] = (unsigned short)(apair >> 16);
            int kb = kbase + 16 * (p >> 2) + half * 8 + 2 * (p & 3);
            bu[2 * p]     = f2bf(Wf[(kb)     * FLEN + n0 + i]);
            bu[2 * p + 1] = f2bf(Wf[(kb + 1) * FLEN + n0 + i]);
        }
        acc = __builtin_amdgcn_wmma_f32_16x16x32_bf16(
                  false, __builtin_bit_cast(v16bf, au),
                  false, __builtin_bit_cast(v16bf, bu),
                  (short)0, acc, false, false);
    }
    float bias = bfv[n0 + i];
#pragma unroll
    for (int v = 0; v < 8; ++v) {
        float t = acc[v] + bias;
        hbf[(b0r + v + 8 * half) * FLEN + n0 + i] = f2bf(t / (1.f + fabsf(t)));
    }
}

// -------------------------------------------------------------------------
// Kernel 5: block-Toeplitz convolution via WMMA, LDS-staged.
// Block = 256 threads = 8 waves = one batch segment of 2048 outputs.
//   - x window (161 blocks, zero-padded to 164) staged to LDS as bf16 once
//   - filter staged to LDS (async DMA to LDS when the builtin exists)
//   - banded Toeplitz B-matrix Bt[j][k]=h[16*(k>>4)+j-(k&15)] built in LDS
//   - inner loop: 2x ds_load_b128 (A) + 2x ds_load_b128 (B) per WMMA
// -------------------------------------------------------------------------
#define XBLKS 164                 // 33 (history) + 128 (segment) + 3 pad
#define KP    544                 // 34 t-blocks * 16

__global__ void conv_wmma_kernel(const float* __restrict__ x,
                                 const unsigned short* __restrict__ hbf,
                                 float* __restrict__ y) {
    __shared__ __align__(16) unsigned short s_lx[XBLKS * 16];   // 5248 B
    __shared__ __align__(16) unsigned short s_lh[576];          // 1152 B (16 pad | 512 | 48 pad)
    __shared__ __align__(16) unsigned short s_Bt[16 * KP];      // 17408 B

    int tid   = threadIdx.x;
    int b     = blockIdx.x >> 4;          // batch
    int seg   = blockIdx.x & 15;          // 16 segments per batch
    int I0seg = seg * 128;                // first block-row of segment
    const float*          xb = x   + (size_t)b * MBS;
    const unsigned short* hb = hbf + b * FLEN;

    // ---- stage filter into s_lh (interior = raw bf16 copy, edges = 0) ----
#ifdef HAVE_ASYNC_LDS
    if (tid < 64) {
        char* gsrc = (char*)hb;
        char* ldst = (char*)(s_lh + 16);
        __builtin_amdgcn_global_load_async_to_lds_b128(
            (gv4i_p)(gsrc + tid * 16),
            (lv4i_p)(ldst + tid * 16), 0, 0);
    } else if (tid < 80) {
        s_lh[tid - 64] = 0;                       // low pad  [0,16)
    } else if (tid < 128) {
        s_lh[528 + (tid - 80)] = 0;               // high pad [528,576)
    }
#else
    for (int e = tid; e < 576; e += 256) {
        int hi = e - 16;
        s_lh[e] = (hi >= 0 && hi < FLEN) ? hb[hi] : (unsigned short)0;
    }
#endif

    // ---- stage x window as bf16, zero-padded ----
    for (int e = tid; e < XBLKS * 16; e += 256) {
        int blk = I0seg - 33 + (e >> 4);
        float v = 0.f;
        if (blk >= 0 && blk < MBS / 16) v = xb[blk * 16 + (e & 15)];
        s_lx[e] = f2bf(v);
    }

#ifdef HAVE_ASYNC_LDS
    __builtin_amdgcn_s_wait_asynccnt(0);
#endif
    __syncthreads();

    // ---- build banded Toeplitz Bt[j][k] = h[16*(k>>4) + j - (k&15)] ----
    {
        int j  = tid >> 4;        // 0..15
        int s0 = tid & 15;        // 0..15
        const unsigned short* lh = s_lh + 16 + j - s0;   // pre-offset
#pragma unroll
        for (int t = 0; t < 34; ++t)
            s_Bt[j * KP + t * 16 + s0] = lh[t * 16];
    }
    __syncthreads();

    // ---- main loop: one 16x16 output tile per wave ----
    int w    = tid >> 5;                  // wave in block: 0..7
    int lane = tid & 31;
    int i = lane & 15, half = lane >> 4, j = i;
    int I0w = I0seg + 16 * w;

    const char* baseA = (const char*)s_lx + (((16 * w + i + 33) * 16 + half * 8) * 2);
    const char* baseB = (const char*)s_Bt + ((j * KP + 8 * half) * 2);

    v8f acc = {};
    for (int c2 = 0; c2 < 17; ++c2) {
        B32B afr, bfr;
        afr.lo = *(const uint4*)(baseA - (2 * c2 + 0) * 32);   // group 0 (t = 2c)
        afr.hi = *(const uint4*)(baseA - (2 * c2 + 1) * 32);   // group 1 (t = 2c+1)
        bfr.lo = *(const uint4*)(baseB + c2 * 64);             // k = 32c+8h+0..7
        bfr.hi = *(const uint4*)(baseB + c2 * 64 + 32);        // k = 32c+16+8h+0..7
        acc = __builtin_amdgcn_wmma_f32_16x16x32_bf16(
                  false, __builtin_bit_cast(v16bf, afr),
                  false, __builtin_bit_cast(v16bf, bfr),
                  (short)0, acc, false, false);
    }

    float* yb = y + (size_t)b * MBS;
#pragma unroll
    for (int v = 0; v < 8; ++v)
        yb[16 * (I0w + v + 8 * half) + j] = acc[v];
}

// -------------------------------------------------------------------------
extern "C" void kernel_launch(void* const* d_in, const int* in_sizes, int n_in,
                              void* d_out, int out_size, void* d_ws, size_t ws_size,
                              hipStream_t stream) {
    const float* inputs = (const float*)d_in[0];
    const float* c      = (const float*)d_in[1];
    const float* W0     = (const float*)d_in[2];
    const float* b0     = (const float*)d_in[3];
    const float* alpha  = (const float*)d_in[4];
    const float* Wg     = (const float*)d_in[5];
    const float* bg     = (const float*)d_in[6];
    const float* Wb     = (const float*)d_in[7];
    const float* bb     = (const float*)d_in[8];
    const float* Wglu   = (const float*)d_in[9];
    const float* bglu   = (const float*)d_in[10];
    const float* Wf     = (const float*)d_in[11];
    const float* bfv    = (const float*)d_in[12];
    float* out = (float*)d_out;

    char* ws = (char*)d_ws;
    unsigned short* zbf  = (unsigned short*)ws;                           // 512 KB
    float*          G    = (float*)(ws + 512 * 1024);                     // 2 MB
    unsigned short* zgbf = (unsigned short*)(ws + 512 * 1024 + 2097152);  // 512 KB
    unsigned short* hbf  = (unsigned short*)(ws + 1024 * 1024 + 2097152); // 512 KB

    cond_film_kernel<<<BATCH, FLEN, 0, stream>>>(c, W0, b0, alpha, Wg, bg, Wb, bb, zbf);
    glu_gemm_kernel <<<(2048 * 32) / 256, 256, 0, stream>>>(zbf, Wglu, G);
    glu_act_kernel  <<<(BATCH * FLEN) / 256, 256, 0, stream>>>(G, bglu, zgbf);
    filt_gemm_kernel<<<(1024 * 32) / 256, 256, 0, stream>>>(zgbf, Wf, bfv, hbf);
    conv_wmma_kernel<<<BATCH * 16, 256, 0, stream>>>(inputs, hbf, out);
}
